// BGNet_7816840479312
// MI455X (gfx1250) — matrix operand order repaired and checked by reading, not compiled, not run on hardware
//
#include <hip/hip_runtime.h>
#include <hip/hip_bf16.h>

// ---------------------------------------------------------------------------
// BGNet head: GWC volume -> conv3d(44->32)+BN+LReLU -> conv3d(32->16)+BN+LReLU
// B=4, C=352, H=96, W=192, G=44, D=25.  WMMA bf16 implicit GEMM for the convs.
// Zero-halo padded intermediates => branch-free fully-unrolled WMMA loops with
// compile-time immediate offsets. N=32 per wave: A fragments amortized over
// two 16-column tiles (2 global b128 loads per WMMA in conv1).
// ---------------------------------------------------------------------------

#define Bn 4
#define Cn 352
#define Hn 96
#define Wn 192
#define Gn 44
#define Dn 25
#define EPSB 1e-5f
#define SLOPE 0.1f

// padded dims (1-element zero halo on D, H, W)
#define Dp (Dn + 2)                       // 27
#define Hp (Hn + 2)                       // 98
#define Wp (Wn + 2)                       // 194
#define PADPOS (Bn * Dp * Hp * Wp)        // 2,053,296 padded positions

#define CHUNKS32 ((Bn * Dn * Hn * Wn) / 32) // 57,600 wave tiles (32 W-positions)
#define CPR32 (Wn / 32)                     // 6 chunks per (b,d,h) row

typedef __attribute__((ext_vector_type(16))) __bf16 v16bf;
typedef __attribute__((ext_vector_type(8)))  float  v8f;

static __device__ __forceinline__ unsigned short f2bf(float f) {
    unsigned int u = __float_as_uint(f);
    unsigned int r = u + 0x7FFFu + ((u >> 16) & 1u);   // round-to-nearest-even
    return (unsigned short)(r >> 16);
}

// A-fragment k index (contiguous order, matching the B loads):
// lane half h (0/1), element i in v16bf:  k = i + 16*h
static __device__ __forceinline__ int a_kmap(int i, int half) {
    return i + 16 * half;
}

// ---------------------------------------------------------------------------
// Zero-fill (halo init for padded vol + act1). Capture-safe (plain kernel).
// ---------------------------------------------------------------------------
__global__ void zero_kernel(uint4* __restrict__ p, long n16) {
    long i = (long)blockIdx.x * blockDim.x + threadIdx.x;
    if (i < n16) {
        uint4 z; z.x = 0u; z.y = 0u; z.z = 0u; z.w = 0u;
        p[i] = z;
    }
}

// ---------------------------------------------------------------------------
// Weight prep: pack w1 [32,44,3,3,3] -> [tap][chunk][mtile][lane][16] bf16
// (Cin padded 44->64 with zeros: 2 K-chunks of 32)
// ---------------------------------------------------------------------------
__global__ void wprep1_kernel(const float* __restrict__ w1,
                              unsigned short* __restrict__ wpack1) {
    int idx = blockIdx.x * blockDim.x + threadIdx.x;
    if (idx >= 27 * 2 * 2 * 32 * 16) return;
    int i     = idx & 15;
    int lane  = (idx >> 4) & 31;
    int mtile = (idx >> 9) & 1;
    int chunk = (idx >> 10) & 1;
    int tap   = idx >> 11;
    int row   = lane & 15;
    int half  = lane >> 4;
    int k     = a_kmap(i, half);
    int cin   = chunk * 32 + k;
    int o     = mtile * 16 + row;
    float v = 0.0f;
    if (cin < Gn) v = w1[((size_t)o * Gn + cin) * 27 + tap];  // OIDHW, tap=kd*9+kh*3+kw
    wpack1[idx] = f2bf(v);
}

// w2 [16,32,3,3,3] -> [tap][lane][16] bf16 (K=32 exact)
__global__ void wprep2_kernel(const float* __restrict__ w2,
                              unsigned short* __restrict__ wpack2) {
    int idx = blockIdx.x * blockDim.x + threadIdx.x;
    if (idx >= 27 * 32 * 16) return;
    int i    = idx & 15;
    int lane = (idx >> 4) & 31;
    int tap  = idx >> 9;
    int row  = lane & 15;
    int half = lane >> 4;
    int cin  = a_kmap(i, half);        // 0..31, all valid
    wpack2[idx] = f2bf(w2[((size_t)row * 32 + cin) * 27 + tap]);
}

// ---------------------------------------------------------------------------
// GWC volume into padded layout: volp[b][d+1][h+1][w+1][64] bf16 (ch 44..63=0).
// Block = 256 threads = 32 W-positions x 8 group-offsets; d in chunks of 5
// through a 20KB LDS tile so global stores are coalesced 16B/thread.
// ---------------------------------------------------------------------------
__global__ void vol_kernel(const float* __restrict__ ref,
                           const float* __restrict__ tgt,
                           unsigned short* __restrict__ volp) {
    __shared__ __align__(16) unsigned short tile[5][32][64];   // 20,480 B
    int bid   = blockIdx.x;
    int wtile = bid % (Wn / 32);
    int h     = (bid / (Wn / 32)) % Hn;
    int b     = bid / ((Wn / 32) * Hn);
    int t     = threadIdx.x;
    int wl    = t & 31;
    int w     = wtile * 32 + wl;
    int gpo   = t >> 5;                 // 0..7

    for (int dc = 0; dc < 5; ++dc) {
        for (int gi = 0; gi < 8; ++gi) {
            int g = gi * 8 + gpo;       // 0..63
            float val[5];
            if (g < Gn) {
                float r[8];
                size_t base = (((size_t)b * Cn + (size_t)g * 8) * Hn + h) * Wn + w;
                for (int c = 0; c < 8; ++c) r[c] = ref[base + (size_t)c * (Hn * Wn)];
                for (int dd = 0; dd < 5; ++dd) {
                    int d  = dc * 5 + dd;
                    int wd = w - d;
                    float s = 0.0f;
                    if (wd >= 0) {
                        size_t tb = (((size_t)b * Cn + (size_t)g * 8) * Hn + h) * Wn + wd;
                        for (int c = 0; c < 8; ++c)
                            s += r[c] * tgt[tb + (size_t)c * (Hn * Wn)];
                        s *= 0.125f;    // mean over 8 channels in group
                    }
                    val[dd] = s;
                }
            } else {
                for (int dd = 0; dd < 5; ++dd) val[dd] = 0.0f;
            }
            for (int dd = 0; dd < 5; ++dd) tile[dd][wl][g] = f2bf(val[dd]);
        }
        __syncthreads();
        // flush 5 d-slices, each 32 pos x 64 ch x 2B = 4KB contiguous
        for (int dd = 0; dd < 5; ++dd) {
            int d = dc * 5 + dd;
            size_t P = (((size_t)b * Dp + (d + 1)) * Hp + (h + 1)) * (size_t)Wp
                     + (size_t)(wtile * 32 + 1);
            uint4 v4 = *(const uint4*)(&tile[dd][0][0] + t * 8);
            *(uint4*)(volp + P * 64 + (size_t)t * 8) = v4;
        }
        __syncthreads();
    }
}

// ---------------------------------------------------------------------------
// conv1: implicit GEMM, M=32 out-ch (2 m-tiles), N=32 W-positions per wave
// (two 16-col tiles sharing A), K = 27 taps x 64 padded ch (2 chunks of 32).
// Fully unrolled, branch-free, constant immediate offsets. BN+LReLU fused.
// ---------------------------------------------------------------------------
__global__ void conv1_kernel(const unsigned short* __restrict__ volp,
                             const unsigned short* __restrict__ wpack1,
                             const float* __restrict__ g1, const float* __restrict__ b1,
                             const float* __restrict__ m1, const float* __restrict__ v1,
                             unsigned short* __restrict__ act1p) {
    __shared__ __align__(32) unsigned short ldsOut[8][1024];   // 8 waves x 2KB
    int wv   = threadIdx.x >> 5;
    int lane = threadIdx.x & 31;
    int half = lane >> 4;
    int n    = lane & 15;

    int cid = blockIdx.x * 8 + wv;          // 0..CHUNKS32-1
    int w0  = (cid % CPR32) * 32;
    int tmp = cid / CPR32;
    int h   = tmp % Hn;  tmp /= Hn;
    int d   = tmp % Dn;
    int b   = tmp / Dn;

    // per-lane base pointer at padded center position (b, d, h, w0+n)
    size_t Pc = (((size_t)b * Dp + (d + 1)) * Hp + (h + 1)) * (size_t)Wp
              + (size_t)(w0 + n + 1);
    const unsigned short* pB = volp + Pc * 64 + half * 16;
    const unsigned short* pA = wpack1 + lane * 16;

    v8f c0a = {};   // mtile 0, cols 0..15
    v8f c1a = {};   // mtile 1, cols 0..15
    v8f c0b = {};   // mtile 0, cols 16..31
    v8f c1b = {};   // mtile 1, cols 16..31

#pragma unroll
    for (int kd = 0; kd < 3; ++kd) {
#pragma unroll
        for (int kh = 0; kh < 3; ++kh) {
#pragma unroll
            for (int kw = 0; kw < 3; ++kw) {
                const int tap  = (kd * 3 + kh) * 3 + kw;
                const int toff = ((kd - 1) * Hp + (kh - 1)) * Wp + (kw - 1); // pos units
#pragma unroll
                for (int chunk = 0; chunk < 2; ++chunk) {
                    v16bf bfa = *(const v16bf*)(pB + (ptrdiff_t)toff * 64 + chunk * 32);
                    v16bf bfb = *(const v16bf*)(pB + (ptrdiff_t)toff * 64 + chunk * 32 + 16 * 64);
                    v16bf a0  = *(const v16bf*)(pA + ((tap * 2 + chunk) * 2 + 0) * 512);
                    v16bf a1  = *(const v16bf*)(pA + ((tap * 2 + chunk) * 2 + 1) * 512);
                    c0a = __builtin_amdgcn_wmma_f32_16x16x32_bf16(false, a0, false, bfa,
                                                                  (short)0, c0a, false, false);
                    c1a = __builtin_amdgcn_wmma_f32_16x16x32_bf16(false, a1, false, bfa,
                                                                  (short)0, c1a, false, false);
                    c0b = __builtin_amdgcn_wmma_f32_16x16x32_bf16(false, a0, false, bfb,
                                                                  (short)0, c0b, false, false);
                    c1b = __builtin_amdgcn_wmma_f32_16x16x32_bf16(false, a1, false, bfb,
                                                                  (short)0, c1b, false, false);
                }
            }
        }
    }

    // BN + LeakyReLU, repack [32 pos][32 ch] bf16 into LDS
#pragma unroll
    for (int mt = 0; mt < 2; ++mt) {
#pragma unroll
        for (int r = 0; r < 8; ++r) {
            int ch = mt * 16 + 8 * half + r;
            float sc = g1[ch] * rsqrtf(v1[ch] + EPSB);
            float mB = b1[ch] - m1[ch] * sc;
            float ya = ((mt == 0) ? c0a[r] : c1a[r]) * sc + mB;
            float yb = ((mt == 0) ? c0b[r] : c1b[r]) * sc + mB;
            ya = (ya >= 0.0f) ? ya : SLOPE * ya;
            yb = (yb >= 0.0f) ? yb : SLOPE * yb;
            ldsOut[wv][n * 32 + ch]         = f2bf(ya);
            ldsOut[wv][(16 + n) * 32 + ch]  = f2bf(yb);
        }
    }
    __syncthreads();
    // wave's 32 positions are contiguous in padded act1 -> 2KB contiguous block
    size_t Ps = (((size_t)b * Dp + (d + 1)) * Hp + (h + 1)) * (size_t)Wp
              + (size_t)(w0 + 1);
    *(v16bf*)(act1p + Ps * 32 + (size_t)lane * 32) =
        *(const v16bf*)(&ldsOut[wv][0] + lane * 32);
    *(v16bf*)(act1p + Ps * 32 + (size_t)lane * 32 + 16) =
        *(const v16bf*)(&ldsOut[wv][0] + lane * 32 + 16);
}

// ---------------------------------------------------------------------------
// conv2: M=16 out-ch, N=32 positions (two 16-col tiles sharing A),
// K = 27 taps x 32 channels. Fully unrolled, branch-free. f32 NCDHW output.
// ---------------------------------------------------------------------------
__global__ void conv2_kernel(const unsigned short* __restrict__ act1p,
                             const unsigned short* __restrict__ wpack2,
                             const float* __restrict__ g2, const float* __restrict__ b2,
                             const float* __restrict__ m2, const float* __restrict__ v2,
                             float* __restrict__ out) {
    int wv   = threadIdx.x >> 5;
    int lane = threadIdx.x & 31;
    int half = lane >> 4;
    int n    = lane & 15;

    int cid = blockIdx.x * 8 + wv;
    int w0  = (cid % CPR32) * 32;
    int tmp = cid / CPR32;
    int h   = tmp % Hn;  tmp /= Hn;
    int d   = tmp % Dn;
    int b   = tmp / Dn;

    size_t Pc = (((size_t)b * Dp + (d + 1)) * Hp + (h + 1)) * (size_t)Wp
              + (size_t)(w0 + n + 1);
    const unsigned short* pB = act1p + Pc * 32 + half * 16;
    const unsigned short* pA = wpack2 + lane * 16;

    v8f ca = {};
    v8f cb = {};
#pragma unroll
    for (int kd = 0; kd < 3; ++kd) {
#pragma unroll
        for (int kh = 0; kh < 3; ++kh) {
#pragma unroll
            for (int kw = 0; kw < 3; ++kw) {
                const int tap  = (kd * 3 + kh) * 3 + kw;
                const int toff = ((kd - 1) * Hp + (kh - 1)) * Wp + (kw - 1);
                v16bf bfa = *(const v16bf*)(pB + (ptrdiff_t)toff * 32);
                v16bf bfb = *(const v16bf*)(pB + (ptrdiff_t)toff * 32 + 16 * 32);
                v16bf a   = *(const v16bf*)(pA + tap * 512);
                ca = __builtin_amdgcn_wmma_f32_16x16x32_bf16(false, a, false, bfa,
                                                             (short)0, ca, false, false);
                cb = __builtin_amdgcn_wmma_f32_16x16x32_bf16(false, a, false, bfb,
                                                             (short)0, cb, false, false);
            }
        }
    }

#pragma unroll
    for (int r = 0; r < 8; ++r) {
        int ch = 8 * half + r;
        float sc = g2[ch] * rsqrtf(v2[ch] + EPSB);
        float mB = b2[ch] - m2[ch] * sc;
        float ya = ca[r] * sc + mB;
        float yb = cb[r] * sc + mB;
        ya = (ya >= 0.0f) ? ya : SLOPE * ya;
        yb = (yb >= 0.0f) ? yb : SLOPE * yb;
        size_t rowb = (((size_t)(b * 16 + ch) * Dn + d) * Hn + h) * (size_t)Wn;
        out[rowb + (size_t)(w0 + n)]      = ya;
        out[rowb + (size_t)(w0 + 16 + n)] = yb;
    }
}

// ---------------------------------------------------------------------------
extern "C" void kernel_launch(void* const* d_in, const int* in_sizes, int n_in,
                              void* d_out, int out_size, void* d_ws, size_t ws_size,
                              hipStream_t stream) {
    const float* ref = (const float*)d_in[0];
    const float* tgt = (const float*)d_in[1];
    const float* w1  = (const float*)d_in[2];
    const float* g1  = (const float*)d_in[3];
    const float* b1  = (const float*)d_in[4];
    const float* m1  = (const float*)d_in[5];
    const float* v1  = (const float*)d_in[6];
    const float* w2  = (const float*)d_in[7];
    const float* g2  = (const float*)d_in[8];
    const float* b2  = (const float*)d_in[9];
    const float* m2  = (const float*)d_in[10];
    const float* v2  = (const float*)d_in[11];
    float* out = (float*)d_out;

    char* ws = (char*)d_ws;
    // volp: PADPOS*64 bf16 = PADPOS*128 B ; act1p: PADPOS*32 bf16 = PADPOS*64 B
    unsigned short* volp   = (unsigned short*)(ws);
    unsigned short* act1p  = (unsigned short*)(ws + (size_t)PADPOS * 128);
    unsigned short* wpack1 = (unsigned short*)(ws + (size_t)PADPOS * 192);
    unsigned short* wpack2 = (unsigned short*)((char*)wpack1 + 27 * 2 * 2 * 32 * 16 * 2);

    // zero vol + act1 (covers the halos; interiors are overwritten below)
    long n16 = ((long)PADPOS * 192) / 16;   // bytes / 16
    zero_kernel<<<(unsigned)((n16 + 255) / 256), 256, 0, stream>>>((uint4*)ws, n16);

    wprep1_kernel<<<(27 * 2 * 2 * 32 * 16 + 255) / 256, 256, 0, stream>>>(w1, wpack1);
    wprep2_kernel<<<(27 * 32 * 16 + 255) / 256, 256, 0, stream>>>(w2, wpack2);
    vol_kernel<<<Bn * Hn * (Wn / 32), 256, 0, stream>>>(ref, tgt, volp);
    conv1_kernel<<<CHUNKS32 / 8, 256, 0, stream>>>(volp, wpack1, g1, b1, m1, v1, act1p);
    conv2_kernel<<<CHUNKS32 / 8, 256, 0, stream>>>(act1p, wpack2, g2, b2, m2, v2, out);
    (void)in_sizes; (void)n_in; (void)out_size; (void)ws_size;
}
